// FAGCN_Net_8323646620244
// MI455X (gfx1250) — matrix operand order repaired and checked by reading, not compiled, not run on hardware
//
#include <hip/hip_runtime.h>
#include <hip/hip_bf16.h>
#include <math.h>

#define FIN  256
#define NHID 64
#define NCLS 10
#define FAGCN_EPS 0.3f

typedef __attribute__((ext_vector_type(16))) __bf16 v16bf;
typedef __attribute__((ext_vector_type(8)))  float  v8f;

__device__ __forceinline__ void atom_add_f32(float* p, float v) {
    __hip_atomic_fetch_add(p, v, __ATOMIC_RELAXED, __HIP_MEMORY_SCOPE_AGENT);
}

// ---------------------------------------------------------------- zero fill
__global__ void k_zero(float* __restrict__ p, long n) {
    long i = (long)blockIdx.x * blockDim.x + threadIdx.x;
    long s = (long)gridDim.x * blockDim.x;
    for (; i < n; i += s) p[i] = 0.0f;
}

// ---------------------------------------------------------------- degree
__global__ void k_deg(const int* __restrict__ dst, float* __restrict__ deg, int E) {
    long e = (long)blockIdx.x * blockDim.x + threadIdx.x;
    if (e < E) atom_add_f32(deg + dst[e], 1.0f);
}

__global__ void k_dis(float* __restrict__ deg, int N) {
    long n = (long)blockIdx.x * blockDim.x + threadIdx.x;
    if (n < N) {
        float d = deg[n];
        deg[n] = (d > 0.0f) ? rsqrtf(fmaxf(d, 1.0f)) : 0.0f;
    }
}

__global__ void k_norm(const int* __restrict__ src, const int* __restrict__ dst,
                       const float* __restrict__ dis, float* __restrict__ norm, int E) {
    long e = (long)blockIdx.x * blockDim.x + threadIdx.x;
    if (e < E) norm[e] = dis[src[e]] * dis[dst[e]];
}

// ---------------------------------------------------------------- GEMM1: raw = relu(x @ w1^T + b1)
// block: 256 thr = 8 waves; block tile 128 rows x 64 cols; wave tile 16 x 64.
// K = 256, stepped 32 at a time with v_wmma_f32_16x16x32_bf16.
__global__ __launch_bounds__(256) void k_gemm1_relu(
        const float* __restrict__ x, const float* __restrict__ w1,
        const float* __restrict__ b1, float* __restrict__ out, int N) {
    __shared__ __align__(32) __bf16 wlds[NHID * FIN];   // 32 KB, bf16 copy of w1

    int tid = threadIdx.x;
    // stage w1 -> LDS (each thread converts 64 contiguous values)
    {
        int base = tid * 64;
        #pragma unroll 8
        for (int j = 0; j < 64; ++j) wlds[base + j] = (__bf16)w1[base + j];
    }
    __syncthreads();

    int lane = tid & 31;
    int wave = tid >> 5;
    int m    = lane & 15;     // row of A; column of B and C
    int hk   = lane >> 4;     // K-half selector (contiguous k = 16*hk + s order used here)

    long rowbase = (long)blockIdx.x * 128 + (long)wave * 16;
    long row = rowbase + m;
    long rowc = (row < N) ? row : (long)(N - 1);   // clamp loads; stores are guarded
    const float* xrow = x + rowc * FIN;

    v8f acc[4] = {v8f{}, v8f{}, v8f{}, v8f{}};

    for (int k0 = 0; k0 < FIN; k0 += 32) {
        // A fragment: 16x32 bf16. lane<16: K = k0 + {0..15};
        //             lane>=16:   K = k0 + {16..31}
        v16bf a;
        const float* ap0 = xrow + k0 + hk * 16;
        const float* ap1 = ap0 + 8;
        #pragma unroll
        for (int j = 0; j < 8; ++j) a[j]     = (__bf16)ap0[j];
        #pragma unroll
        for (int j = 0; j < 8; ++j) a[8 + j] = (__bf16)ap1[j];

        // B fragments from LDS: col-tile t, lane holds col = 16t + (lane&15),
        // K = k0 + hk*16 + {0..15}  (16 contiguous bf16 = 32B, aligned)
        #pragma unroll
        for (int t = 0; t < 4; ++t) {
            const __bf16* bp = &wlds[(t * 16 + m) * FIN + k0 + hk * 16];
            v16bf b = *(const v16bf*)bp;
            acc[t] = __builtin_amdgcn_wmma_f32_16x16x32_bf16(
                false, a, false, b, (short)0, acc[t], false, false);
        }
    }

    // epilogue: C layout -> lane (l&15) = col within tile, VGPR r = row r + 8*hk
    #pragma unroll
    for (int t = 0; t < 4; ++t) {
        int col = t * 16 + m;
        float bias = b1[col];
        #pragma unroll
        for (int r = 0; r < 8; ++r) {
            long rr = rowbase + r + hk * 8;
            if (rr < N) {
                float v = acc[t][r] + bias;
                out[rr * NHID + col] = v > 0.0f ? v : 0.0f;
            }
        }
    }
}

// ---------------------------------------------------------------- per-node attention scores
// one wave (32 lanes) per node; lane covers channels {l, l+32}; shuffle-reduce.
__global__ __launch_bounds__(256) void k_attn(
        const float* __restrict__ h, const float* __restrict__ attl,
        const float* __restrict__ attr, float* __restrict__ al,
        float* __restrict__ ar, int N) {
    int lane = threadIdx.x & 31;
    long n = (long)blockIdx.x * 8 + (threadIdx.x >> 5);
    if (n >= N) return;
    const float* hr = h + n * NHID;
    float h0 = hr[lane], h1 = hr[32 + lane];
    float sl = h0 * attl[lane] + h1 * attl[32 + lane];
    float sr = h0 * attr[lane] + h1 * attr[32 + lane];
    #pragma unroll
    for (int off = 16; off > 0; off >>= 1) {
        sl += __shfl_xor(sl, off, 32);
        sr += __shfl_xor(sr, off, 32);
    }
    if (lane == 0) { al[n] = sl; ar[n] = sr; }
}

// ---------------------------------------------------------------- edge scatter (atomics into L2)
// one wave per edge; lane covers channels {l, l+32}.
__global__ __launch_bounds__(256) void k_edge(
        const int* __restrict__ src, const int* __restrict__ dst,
        const float* __restrict__ al, const float* __restrict__ ar,
        const float* __restrict__ norm, const float* __restrict__ h,
        float* __restrict__ agg, int E) {
    int lane = threadIdx.x & 31;
    long e = (long)blockIdx.x * 8 + (threadIdx.x >> 5);
    if (e >= E) return;
    int s = src[e], d = dst[e];
    float a = tanhf(al[s] + ar[d]) * norm[e];
    const float* hs = h + (long)s * NHID;
    float* ad = agg + (long)d * NHID;
    atom_add_f32(ad + lane,      hs[lane]      * a);
    atom_add_f32(ad + 32 + lane, hs[32 + lane] * a);
}

// ---------------------------------------------------------------- h = agg + eps*raw
__global__ void k_combine(const float* __restrict__ agg, const float* __restrict__ raw,
                          float* __restrict__ out, long n) {
    long i = (long)blockIdx.x * blockDim.x + threadIdx.x;
    if (i < n) out[i] = agg[i] + FAGCN_EPS * raw[i];
}

// ---------------------------------------------------------------- head: logits + log_softmax
__global__ __launch_bounds__(256) void k_head(
        const float* __restrict__ h, const float* __restrict__ w2,
        const float* __restrict__ b2, float* __restrict__ out, int N) {
    __shared__ float w2s[NCLS * NHID];
    __shared__ float b2s[NCLS];
    int tid = threadIdx.x;
    for (int i = tid; i < NCLS * NHID; i += 256) w2s[i] = w2[i];
    if (tid < NCLS) b2s[tid] = b2[tid];
    __syncthreads();

    long n = (long)blockIdx.x * 256 + tid;
    if (n >= N) return;
    const float* hr = h + n * NHID;
    float hv[NHID];
    #pragma unroll
    for (int k = 0; k < NHID; k += 4) {
        float4 v = *(const float4*)(hr + k);
        hv[k] = v.x; hv[k+1] = v.y; hv[k+2] = v.z; hv[k+3] = v.w;
    }
    float lg[NCLS];
    float mx = -INFINITY;
    #pragma unroll
    for (int c = 0; c < NCLS; ++c) {
        float s = b2s[c];
        #pragma unroll
        for (int k = 0; k < NHID; ++k) s = fmaf(hv[k], w2s[c * NHID + k], s);
        lg[c] = s;
        mx = fmaxf(mx, s);
    }
    float se = 0.0f;
    #pragma unroll
    for (int c = 0; c < NCLS; ++c) se += __expf(lg[c] - mx);
    float lse = mx + __logf(se);
    float* op = out + n * NCLS;
    #pragma unroll
    for (int c = 0; c < NCLS; ++c) op[c] = lg[c] - lse;
}

// ---------------------------------------------------------------- launcher
extern "C" void kernel_launch(void* const* d_in, const int* in_sizes, int n_in,
                              void* d_out, int out_size, void* d_ws, size_t ws_size,
                              hipStream_t stream) {
    const float* x     = (const float*)d_in[0];
    const int*   eidx  = (const int*)  d_in[1];
    const float* w1    = (const float*)d_in[2];
    const float* b1    = (const float*)d_in[3];
    const float* w2    = (const float*)d_in[4];
    const float* b2    = (const float*)d_in[5];
    const float* att_l = (const float*)d_in[6];
    const float* att_r = (const float*)d_in[7];
    float*       out   = (float*)d_out;

    const int N = in_sizes[0] / FIN;
    const int E = in_sizes[1] / 2;
    const int* src = eidx;
    const int* dst = eidx + E;

    // workspace layout (floats)
    float* ws   = (float*)d_ws;
    long   nh   = (long)N * NHID;
    float* raw  = ws;              // N*64
    float* hcur = raw  + nh;       // N*64
    float* agg  = hcur + nh;       // N*64
    float* deg  = agg  + nh;       // N   (becomes dis in place)
    float* al   = deg  + N;        // N
    float* ar   = al   + N;        // N
    float* norm = ar   + N;        // E

    dim3 blk(256);
    int gE256 = (E + 255) / 256;
    int gN256 = (N + 255) / 256;
    int gE8   = (E + 7) / 8;       // wave-per-edge
    int gN8   = (N + 7) / 8;       // wave-per-node
    int gGemm = (N + 127) / 128;

    // degrees -> dis -> per-edge norm
    k_zero<<<512, blk, 0, stream>>>(deg, (long)N);
    k_deg<<<gE256, blk, 0, stream>>>(dst, deg, E);
    k_dis<<<gN256, blk, 0, stream>>>(deg, N);
    k_norm<<<gE256, blk, 0, stream>>>(src, dst, deg, norm, E);

    // raw = relu(x @ w1^T + b1)   (WMMA bf16)
    k_gemm1_relu<<<gGemm, blk, 0, stream>>>(x, w1, b1, raw, N);

    // FAConv layers
    const float* hin = raw;
    for (int l = 0; l < 2; ++l) {
        k_zero<<<2048, blk, 0, stream>>>(agg, nh);
        k_attn<<<gN8, blk, 0, stream>>>(hin, att_l + l * NHID, att_r + l * NHID, al, ar, N);
        k_edge<<<gE8, blk, 0, stream>>>(src, dst, al, ar, norm, hin, agg, E);
        k_combine<<<(int)((nh + 255) / 256), blk, 0, stream>>>(agg, raw, hcur, nh);
        hin = hcur;
    }

    // logits + log_softmax
    k_head<<<gN256, blk, 0, stream>>>(hcur, w2, b2, out, N);
}
